// SparseBPNeuralNetwork_23536420782816
// MI455X (gfx1250) — matrix operand order repaired and checked by reading, not compiled, not run on hardware
//
#include <hip/hip_runtime.h>

// ---------------------------------------------------------------------------
// LDPC BP neural decoder for MI455X (gfx1250, wave32, WMMA).
// All matmuls run through v_wmma_f32_16x16x32_f16 with f32 accumulation.
// Padded dims: N 648->704, E 1944->1984. cn_update's three log/sign-domain
// reductions are stacked into one 6144-row GEMM sharing the mask operand.
// Tile copies use GLOBAL_LOAD_ASYNC_TO_LDS_B128 (ASYNCcnt) double-buffered.
// ---------------------------------------------------------------------------

#ifndef USE_ASYNC_LDS
#define USE_ASYNC_LDS 1   // set 0 to fall back to VGPR-staged ds_store path
#endif

typedef __attribute__((ext_vector_type(16))) _Float16 v16h;
typedef __attribute__((ext_vector_type(8)))  float    v8f;

#define BM 256
#define BN 64
#define BK 32
#define GT 256   // 8 wave32s

__device__ __forceinline__ unsigned short f2h(float f) {
  union { _Float16 h; unsigned short s; } u;
  u.h = (_Float16)f;
  return u.s;
}

union AFrag { v16h v; uint2 q[4]; };
union BFrag { v16h v; uint4 q[2]; };

#if USE_ASYNC_LDS
// one 16-byte async global->LDS copy per lane (ASYNCcnt++)
__device__ __forceinline__ void async_cp16(unsigned lds, const void* g) {
  asm volatile("global_load_async_to_lds_b128 %0, %1, off"
               :: "v"(lds), "v"((unsigned long long)(uintptr_t)g)
               : "memory");
}
#define WAIT_ASYNC_5() asm volatile("s_wait_asynccnt 0x5" ::: "memory")
#define WAIT_ASYNC_0() asm volatile("s_wait_asynccnt 0x0" ::: "memory")
#endif

// C = A[M,K] * B[Nn,K]^T  (NT GEMM, f16 in / f32 accum), fused epilogues.
// mode 0: outF[m*outld+n] = c
// mode 1: n<realN: outF[m*realN+n] = sigmoid(c + aux[m*auxld+n])  (direct d_out)
// mode 2: th = tanh(0.5*(c + aux[m*auxld+n])); write cn_update operand slabs
//         (logmag / neg / nz) as f16 into outS (+0/+slab/+2*slab), ld = outld
__global__ __launch_bounds__(GT)
void gemm_nt_f16(const unsigned short* __restrict__ A,
                 const unsigned short* __restrict__ Bm,
                 int M, int Nn, int K, int mode,
                 const float* __restrict__ aux, int auxld,
                 float* __restrict__ outF, int outld,
                 unsigned short* __restrict__ outS, long long slabStride,
                 int realN)
{
  __shared__ unsigned short sA[2][BM * BK];   // 32 KB
  __shared__ unsigned short sB[2][BN * BK];   //  8 KB

  const int tid  = threadIdx.x;
  const int lane = tid & 31;
  const int wid  = tid >> 5;
  const int wm   = (wid & 3) * 64;   // 4 waves over M, 64 rows each
  const int wn   = (wid >> 2) * 32;  // 2 waves over N, 32 cols each
  const int m0   = blockIdx.y * BM;
  const int n0   = blockIdx.x * BN;

#if USE_ASYNC_LDS
  const unsigned sAb[2] = {(unsigned)(uintptr_t)(void*)&sA[0][0],
                           (unsigned)(uintptr_t)(void*)&sA[1][0]};
  const unsigned sBb[2] = {(unsigned)(uintptr_t)(void*)&sB[0][0],
                           (unsigned)(uintptr_t)(void*)&sB[1][0]};
  auto issueAsync = [&](int buf, int k0) {
#pragma unroll
    for (int i = 0; i < 4; ++i) {               // A tile: 256 rows x 4 segs
      int idx = tid + i * GT;
      int row = idx >> 2, seg = idx & 3;
      async_cp16(sAb[buf] + (unsigned)((row * BK + seg * 8) * 2),
                 A + (size_t)(m0 + row) * K + k0 + seg * 8);
    }
    int row = tid >> 2, seg = tid & 3;          // B tile: 64 rows x 4 segs
    async_cp16(sBb[buf] + (unsigned)((row * BK + seg * 8) * 2),
               Bm + (size_t)(n0 + row) * K + k0 + seg * 8);
  };
#else
  uint4 ra[4], rb;
  auto loadRegs = [&](int k0) {
#pragma unroll
    for (int i = 0; i < 4; ++i) {
      int idx = tid + i * GT;
      int row = idx >> 2, seg = idx & 3;
      ra[i] = *(const uint4*)(A + (size_t)(m0 + row) * K + k0 + seg * 8);
    }
    int row = tid >> 2, seg = tid & 3;
    rb = *(const uint4*)(Bm + (size_t)(n0 + row) * K + k0 + seg * 8);
  };
  auto storeLDS = [&](int buf) {
#pragma unroll
    for (int i = 0; i < 4; ++i) {
      int idx = tid + i * GT;
      int row = idx >> 2, seg = idx & 3;
      *(uint4*)&sA[buf][row * BK + seg * 8] = ra[i];
    }
    int row = tid >> 2, seg = tid & 3;
    *(uint4*)&sB[buf][row * BK + seg * 8] = rb;
  };
#endif

  v8f acc[4][2];
  const v8f vzero = {0.f, 0.f, 0.f, 0.f, 0.f, 0.f, 0.f, 0.f};
#pragma unroll
  for (int mt = 0; mt < 4; ++mt) { acc[mt][0] = vzero; acc[mt][1] = vzero; }

  const int nk  = K / BK;
  const int r16 = lane & 15;
  const int hi  = lane >> 4;

#if USE_ASYNC_LDS
  issueAsync(0, 0);
#else
  loadRegs(0);
  storeLDS(0);
  __syncthreads();
#endif

  for (int kt = 0; kt < nk; ++kt) {
    const int cur = kt & 1;
#if USE_ASYNC_LDS
    if (kt + 1 < nk) {
      issueAsync(cur ^ 1, (kt + 1) * BK);   // 5 new async ops target other buf
      WAIT_ASYNC_5();                        // oldest 5 (cur's tile) complete
    } else {
      WAIT_ASYNC_0();
    }
    __syncthreads();                         // cur tile visible to all waves
#else
    if (kt + 1 < nk) loadRegs((kt + 1) * BK);
#endif

    const unsigned short* sAc = sA[cur];
    const unsigned short* sBc = sB[cur];

    // A fragment (16x32 f16): row = l%16; K-base = (l<16?0:8); per ISA 7.12.2
    AFrag af[4];
    const int kbA = hi * 8;
#pragma unroll
    for (int mt = 0; mt < 4; ++mt) {
      const unsigned short* arow = sAc + (wm + mt * 16 + r16) * BK;
      af[mt].q[0] = *(const uint2*)(arow + kbA);
      af[mt].q[1] = *(const uint2*)(arow + kbA + 4);
      af[mt].q[2] = *(const uint2*)(arow + kbA + 16);
      af[mt].q[3] = *(const uint2*)(arow + kbA + 20);
    }
    // B fragment (32x16 f16): col = l%16, K-base = (l<16?0:8); per ISA 7.12.2
    BFrag bfr[2];
    const int kbB = hi * 8;
#pragma unroll
    for (int nt = 0; nt < 2; ++nt) {
      const unsigned short* brow = sBc + (wn + nt * 16 + r16) * BK;
      bfr[nt].q[0] = *(const uint4*)(brow + kbB);
      bfr[nt].q[1] = *(const uint4*)(brow + kbB + 16);
    }

#pragma unroll
    for (int mt = 0; mt < 4; ++mt)
#pragma unroll
      for (int nt = 0; nt < 2; ++nt)
        acc[mt][nt] = __builtin_amdgcn_wmma_f32_16x16x32_f16(
            false, af[mt].v, false, bfr[nt].v, (short)0, acc[mt][nt],
            false, false);

    __syncthreads();   // all waves done reading cur before it is overwritten
#if !USE_ASYNC_LDS
    if (kt + 1 < nk) { storeLDS(cur ^ 1); __syncthreads(); }
#endif
  }

  // Epilogue. C/D layout: lane col = l%16; VGPR r -> row r (+8 for lanes 16-31)
#pragma unroll
  for (int mt = 0; mt < 4; ++mt) {
#pragma unroll
    for (int nt = 0; nt < 2; ++nt) {
      const int ng = n0 + wn + nt * 16 + r16;
#pragma unroll
      for (int r = 0; r < 8; ++r) {
        const int mg = m0 + wm + mt * 16 + r + hi * 8;
        const float c = acc[mt][nt][r];
        if (mode == 0) {
          outF[(size_t)mg * outld + ng] = c;
        } else if (mode == 1) {
          if (ng < realN) {
            float v = c + aux[(size_t)mg * auxld + ng];
            outF[(size_t)mg * realN + ng] = 1.f / (1.f + __expf(-v));
          }
        } else {
          float th  = tanhf(0.5f * (c + aux[(size_t)mg * auxld + ng]));
          float lm  = (th != 0.f) ? __logf(fabsf(th)) : 0.f;
          float ngv = (th < 0.f) ? 1.f : 0.f;
          float nzv = (th != 0.f) ? 1.f : 0.f;
          size_t o = (size_t)mg * outld + ng;
          outS[o]                  = f2h(lm);
          outS[o + slabStride]     = f2h(ngv);
          outS[o + 2 * slabStride] = f2h(nzv);
        }
      }
    }
  }
}

// f32 -> padded f16 (zero pads)
__global__ void cvt_pad(const float* __restrict__ src, unsigned short* __restrict__ dst,
                        int R, int C, int Rp, int Cp)
{
  int idx = blockIdx.x * GT + threadIdx.x;
  if (idx >= Rp * Cp) return;
  int r = idx / Cp, c = idx % Cp;
  float v = (r < R && c < C) ? src[(size_t)r * C + c] : 0.f;
  dst[idx] = f2h(v);
}

// dst[r][c] = src[c][r], padded (used for bias_matrix^T)
__global__ void cvt_pad_T(const float* __restrict__ src, unsigned short* __restrict__ dst,
                          int Rs, int Cs, int Rp, int Cp)
{
  int idx = blockIdx.x * GT + threadIdx.x;
  if (idx >= Rp * Cp) return;
  int r = idx / Cp, c = idx % Cp;
  float v = (r < Cs && c < Rs) ? src[(size_t)c * Cs + r] : 0.f;
  dst[idx] = f2h(v);
}

// t0 = tanh(0.5*x); write llr (f32 padded + f16 padded) and cn_update slabs
__global__ void prep0_kernel(const float* __restrict__ x,
                             float* __restrict__ llrP,
                             unsigned short* __restrict__ llrB,
                             unsigned short* __restrict__ A3,
                             int Bt, int N, int NP, long long slab)
{
  int idx = blockIdx.x * GT + threadIdx.x;
  if (idx >= Bt * NP) return;
  int r = idx / NP, c = idx % NP;
  float xv = (c < N) ? x[(size_t)r * N + c] : 0.f;
  llrP[idx] = xv;
  llrB[idx] = f2h(xv);
  float th = tanhf(0.5f * xv);
  float lm = (th != 0.f) ? __logf(fabsf(th)) : 0.f;
  A3[idx]            = f2h(lm);
  A3[idx + slab]     = f2h((th < 0.f) ? 1.f : 0.f);
  A3[idx + 2 * slab] = f2h((th != 0.f) ? 1.f : 0.f);
}

// combine cn_update slabs: t = 2*arctanh(clip(sign*exp(logprod))), write f16
__global__ void cn_combine(const float* __restrict__ C3,
                           unsigned short* __restrict__ tbf,
                           int total, long long slab)
{
  int idx = blockIdx.x * GT + threadIdx.x;
  if (idx >= total) return;
  float lp  = C3[idx];
  float nc  = C3[idx + slab];
  float anz = C3[idx + 2 * slab];
  float sg  = 1.f - 2.f * fmodf(nc, 2.f);
  float t   = (anz > 0.5f) ? sg * __expf(lp) : 0.f;
  t = fminf(0.999999f, fmaxf(-0.999999f, t));
  tbf[idx] = f2h(__logf((1.f + t) / (1.f - t)));
}

// ---------------------------------------------------------------------------

extern "C" void kernel_launch(void* const* d_in, const int* in_sizes, int n_in,
                              void* d_out, int out_size, void* d_ws, size_t ws_size,
                              hipStream_t stream)
{
  (void)in_sizes; (void)n_in; (void)out_size; (void)ws_size;
  const int N = 648, E = 1944, BATCH = 2048;
  const int NP = 704, EP = 1984;              // padded to multiples of 64
  const long long SLAB_N = (long long)BATCH * NP;
  const long long SLAB_E = (long long)BATCH * EP;
  const size_t OUTS = (size_t)BATCH * N;

  const float* x      = (const float*)d_in[0];
  const float* Mfirst = (const float*)d_in[1];
  const float* Mcn    = (const float*)d_in[2];
  const float* Mout   = (const float*)d_in[3];
  const float* biasM  = (const float*)d_in[4];
  const float* W[4]   = {(const float*)d_in[5], (const float*)d_in[6],
                         (const float*)d_in[7], (const float*)d_in[8]};
  const float* W9     = (const float*)d_in[9];
  float* out = (float*)d_out;

  char* ws = (char*)d_ws;
  size_t off = 0;
  auto alloc = [&](size_t bytes) -> void* {
    void* p = ws + off;
    off += (bytes + 255) & ~(size_t)255;
    return p;
  };
  unsigned short* Mfirst_h = (unsigned short*)alloc((size_t)EP * NP * 2);
  unsigned short* Mcn_h    = (unsigned short*)alloc((size_t)EP * EP * 2);
  unsigned short* W_h[4];
  for (int i = 0; i < 4; ++i) W_h[i] = (unsigned short*)alloc((size_t)EP * EP * 2);
  unsigned short* Mout_h   = (unsigned short*)alloc((size_t)NP * EP * 2);
  unsigned short* W9_h     = (unsigned short*)alloc((size_t)NP * EP * 2);
  unsigned short* biasT_h  = (unsigned short*)alloc((size_t)EP * NP * 2);
  unsigned short* llr_h    = (unsigned short*)alloc((size_t)BATCH * NP * 2);
  float*          llrP     = (float*)alloc((size_t)BATCH * NP * 4);
  float*          biasTerm = (float*)alloc((size_t)BATCH * EP * 4);
  unsigned short* A3       = (unsigned short*)alloc((size_t)3 * BATCH * EP * 2);
  float*          C3       = (float*)alloc((size_t)3 * BATCH * EP * 4);
  unsigned short* t_h      = (unsigned short*)alloc((size_t)BATCH * EP * 2);

  auto ew = [&](size_t total) { return dim3((unsigned)((total + GT - 1) / GT)); };

  // weight conversion (f32 -> padded f16); masks are 0/1 so exact
  cvt_pad<<<ew((size_t)EP * NP), GT, 0, stream>>>(Mfirst, Mfirst_h, E, N, EP, NP);
  cvt_pad<<<ew((size_t)EP * EP), GT, 0, stream>>>(Mcn, Mcn_h, E, E, EP, EP);
  for (int i = 0; i < 4; ++i)
    cvt_pad<<<ew((size_t)EP * EP), GT, 0, stream>>>(W[i], W_h[i], E, E, EP, EP);
  cvt_pad<<<ew((size_t)NP * EP), GT, 0, stream>>>(Mout, Mout_h, N, E, NP, EP);
  cvt_pad<<<ew((size_t)NP * EP), GT, 0, stream>>>(W9, W9_h, N, E, NP, EP);
  cvt_pad_T<<<ew((size_t)EP * NP), GT, 0, stream>>>(biasM, biasT_h, N, E, EP, NP);

  // stage 0 prep: llr + first cn_update operands
  prep0_kernel<<<ew((size_t)BATCH * NP), GT, 0, stream>>>(
      x, llrP, llr_h, A3, BATCH, N, NP, SLAB_N);

  auto gemm = [&](const unsigned short* A, const unsigned short* Bmat,
                  int M, int Nn, int K, int mode,
                  const float* aux, int auxld,
                  float* outF, int outld,
                  unsigned short* outS, long long slab, int realN) {
    dim3 grid((unsigned)(Nn / BN), (unsigned)(M / BM));
    gemm_nt_f16<<<grid, GT, 0, stream>>>(A, Bmat, M, Nn, K, mode, aux, auxld,
                                         outF, outld, outS, slab, realN);
  };

  // biasTerm = llr @ bias_matrix  (NT with bias^T)  [2048,1984]
  gemm(llr_h, biasT_h, BATCH, EP, NP, 0, nullptr, 0, biasTerm, EP, nullptr, 0, 0);

  // iteration 0: cn_update with M_first, then out1
  gemm(A3, Mfirst_h, 3 * BATCH, EP, NP, 0, nullptr, 0, C3, EP, nullptr, 0, 0);
  cn_combine<<<ew((size_t)BATCH * EP), GT, 0, stream>>>(C3, t_h, BATCH * EP, SLAB_E);
  gemm(t_h, Mout_h, BATCH, NP, EP, 1, llrP, NP, out + 4 * OUTS, 0, nullptr, 0, N);

  // iterations 1..4 (W1, W3, W5, W7)
  for (int i = 0; i < 4; ++i) {
    gemm(t_h, W_h[i], BATCH, EP, EP, 2, biasTerm, EP, nullptr, EP, A3, SLAB_E, 0);
    gemm(A3, Mcn_h, 3 * BATCH, EP, EP, 0, nullptr, 0, C3, EP, nullptr, 0, 0);
    cn_combine<<<ew((size_t)BATCH * EP), GT, 0, stream>>>(C3, t_h, BATCH * EP, SLAB_E);
    if (i < 3) {
      gemm(t_h, Mout_h, BATCH, NP, EP, 1, llrP, NP, out + (size_t)(3 - i) * OUTS,
           0, nullptr, 0, N);
    } else {
      gemm(t_h, W9_h, BATCH, NP, EP, 1, llrP, NP, out, 0, nullptr, 0, N);
    }
  }
}
